// NezhaSelfAttention_74259984547972
// MI455X (gfx1250) — matrix operand order repaired and checked by reading, not compiled, not run on hardware
//
#include <hip/hip_runtime.h>
#include <math.h>

#define Bb 2
#define Ss 1024
#define Hh 1024
#define NHh 16
#define Dd 64
#define VOCAB 129
#define VPAD 144   // vocab padded to multiple of 16 for WMMA N tiles

typedef __attribute__((ext_vector_type(16))) __bf16          v16bf;
typedef __attribute__((ext_vector_type(16))) unsigned short  v16us;
typedef __attribute__((ext_vector_type(8)))  float           v8f;

__device__ __forceinline__ unsigned short f32_to_bf16(float f) {
  union { float f; unsigned u; } x; x.f = f;
  unsigned r = x.u + 0x7FFFu + ((x.u >> 16) & 1u);   // round-to-nearest-even
  return (unsigned short)(r >> 16);
}

__device__ __forceinline__ v8f wmma_bf16(v16us a, v16us b, v8f c) {
  return __builtin_amdgcn_wmma_f32_16x16x32_bf16(
      false, __builtin_bit_cast(v16bf, a),
      false, __builtin_bit_cast(v16bf, b),
      (short)0, c, false, false);
}

// ---------------- fp32 -> bf16 conversion ----------------
__global__ void cvt_f32_bf16(const float* __restrict__ src,
                             unsigned short* __restrict__ dst, int n) {
  int stride = gridDim.x * blockDim.x;
  for (int i = blockIdx.x * blockDim.x + threadIdx.x; i < n; i += stride)
    dst[i] = f32_to_bf16(src[i]);
}

// ---------------- sinusoidal relative-position table ----------------
__global__ void build_tab(float* __restrict__ tabf,
                          unsigned short* __restrict__ tabbf) {
  int tid = blockIdx.x * blockDim.x + threadIdx.x;
  int stride = gridDim.x * blockDim.x;
  for (int i = tid; i < VPAD * Dd; i += stride) {
    int p = i / Dd, d = i % Dd;
    float val = 0.f;
    if (p < VOCAB) {
      float j = (float)(d & ~1);
      float div = __expf(j * (-9.210340371976184f / (float)Dd)); // -ln(10000)/D
      float ang = (float)p * div;
      val = (d & 1) ? __cosf(ang) : __sinf(ang);
      tabf[p * Dd + d] = val;
    }
    tabbf[i] = f32_to_bf16(val);
  }
}

// ---------------- QKV projection GEMM (bf16 WMMA) ----------------
// One wave per 32x64 output tile; ping-pong double-buffered fragments
// (unroll-by-2, no inter-buffer register copies).
// Q,K written as [B][NH][S][D]; V written transposed [B][NH][D][S].
__global__ __launch_bounds__(128) void qkv_gemm(
    const unsigned short* __restrict__ hid,
    const unsigned short* __restrict__ wq,
    const unsigned short* __restrict__ wk,
    const unsigned short* __restrict__ wv,
    const float* __restrict__ bq,
    const float* __restrict__ bk,
    const float* __restrict__ bv,
    unsigned short* __restrict__ qkv) {
  const int lane = threadIdx.x & 31;
  const int wave = threadIdx.x >> 5;
  const int lo = lane & 15, hi = lane >> 4;

  const int z = blockIdx.z;
  const unsigned short* W = (z == 0) ? wq : (z == 1) ? wk : wv;
  const float* bias = (z == 0) ? bq : (z == 1) ? bk : bv;
  unsigned short* dst = qkv + (size_t)z * (Bb * NHh * Ss * Dd);

  int tile = blockIdx.x * 4 + wave;   // 0..1023
  int mt = tile >> 4;                 // 0..63   (M = 2048 / 32)
  int nt = tile & 15;                 // 0..15   (N = 1024 / 64)
  int m0 = mt * 32, n0 = nt * 64;

  const unsigned short* arow0 = hid + (size_t)(m0 + lo) * Hh;
  const unsigned short* arow1 = hid + (size_t)(m0 + 16 + lo) * Hh;

  auto loadA = [&](v16us a[2], int kc) {
#pragma unroll
    for (int e = 0; e < 16; ++e) {
      int k = kc * 32 + hi * 16 + e;
      a[0][e] = arow0[k];
      a[1][e] = arow1[k];
    }
  };
  auto loadB = [&](v16us b[4], int kc) {
#pragma unroll
    for (int n = 0; n < 4; ++n) {
      const unsigned short* brow = W + (size_t)(n0 + n * 16 + lo) * Hh + kc * 32 + hi * 16;
#pragma unroll
      for (int e = 0; e < 16; ++e) b[n][e] = brow[e];
    }
  };

  constexpr int KC = Hh / 32;   // 32 (even)
  v8f acc[2][4] = {};
  v16us a0[2], b0[4], a1[2], b1[4];
  loadA(a0, 0);
  loadB(b0, 0);
  for (int kc = 0; kc < KC; kc += 2) {
    loadA(a1, kc + 1);
    loadB(b1, kc + 1);
#pragma unroll
    for (int mi = 0; mi < 2; ++mi)
#pragma unroll
      for (int n = 0; n < 4; ++n)
        acc[mi][n] = wmma_bf16(a0[mi], b0[n], acc[mi][n]);
    if (kc + 2 < KC) {
      loadA(a0, kc + 2);
      loadB(b0, kc + 2);
    }
#pragma unroll
    for (int mi = 0; mi < 2; ++mi)
#pragma unroll
      for (int n = 0; n < 4; ++n)
        acc[mi][n] = wmma_bf16(a1[mi], b1[n], acc[mi][n]);
  }

#pragma unroll
  for (int mi = 0; mi < 2; ++mi)
#pragma unroll
    for (int n = 0; n < 4; ++n) {
      int col = n0 + n * 16 + lo;
      int hh = col >> 6, dd = col & 63;
      float bval = bias[col];
#pragma unroll
      for (int r = 0; r < 8; ++r) {
        int m = m0 + mi * 16 + r + hi * 8;
        int b = m >> 10, s = m & 1023;
        unsigned short v = f32_to_bf16(acc[mi][n][r] + bval);
        if (z == 2)  // V transposed: [B][NH][D][S]
          dst[(((size_t)b * NHh + hh) * Dd + dd) * Ss + s] = v;
        else         // Q, K: [B][NH][S][D]
          dst[(((size_t)b * NHh + hh) * Ss + s) * Dd + dd] = v;
      }
    }
}

// ---------------- fused attention ----------------
// 4 waves (128 threads) cooperate on one (b, h, 16-q-row) tile.
// Waves split: K-range (scores), u-tiles (q.tab^T), D-tiles (P@V).
__global__ __launch_bounds__(128) void attn(
    const unsigned short* __restrict__ qh,
    const unsigned short* __restrict__ kh,
    const unsigned short* __restrict__ vt,   // [B][NH][D][S]
    const float* __restrict__ tabf,
    const unsigned short* __restrict__ tabbf,
    const float* __restrict__ mask,
    float* __restrict__ out) {
  __shared__ float s_scores[16][Ss];   // 64 KB
  __shared__ float s_qtab[16][VPAD];   // 9 KB
  __shared__ float s_pb[16][VPAD];     // 9 KB
  __shared__ float s_ctx[16][Dd];      // 4 KB
  __shared__ float s_red[16][8];       // 0.5 KB softmax tree reduce

  const int tid = threadIdx.x;
  const int lane = tid & 31;
  const int wave = tid >> 5;
  const int lo = lane & 15, hi = lane >> 4;

  int bid = blockIdx.x;
  int qt = bid & 63;
  int h = (bid >> 6) & 15;
  int b = bid >> 10;
  int q0 = qt * 16;

  const size_t headoff = ((size_t)(b * NHh + h)) * Ss * Dd;
  const unsigned short* Q = qh + headoff;
  const unsigned short* K = kh + headoff;
  const unsigned short* V = vt + headoff;   // transposed head

  for (int i = tid; i < 16 * VPAD; i += 128) ((float*)s_pb)[i] = 0.f;

  // Q fragments (each wave keeps its own copy; same 16 rows)
  v16us qa[2];
#pragma unroll
  for (int c = 0; c < 2; ++c)
#pragma unroll
    for (int e = 0; e < 16; ++e) {
      int d = c * 32 + hi * 16 + e;
      qa[c][e] = Q[(size_t)(q0 + lo) * Dd + d];
    }

  // qtab = Q @ tab^T; waves split the 9 u-tiles
  for (int ut = wave; ut < VPAD / 16; ut += 4) {
    v8f acc = {};
    int ucol = ut * 16 + lo;
#pragma unroll
    for (int c = 0; c < 2; ++c) {
      const unsigned short* trow = tabbf + (size_t)ucol * Dd + c * 32 + hi * 16;
      v16us bfrag;
#pragma unroll
      for (int e = 0; e < 16; ++e) bfrag[e] = trow[e];
      acc = wmma_bf16(qa[c], bfrag, acc);
    }
#pragma unroll
    for (int r = 0; r < 8; ++r) s_qtab[r + hi * 8][ucol] = acc[r];
  }
  __syncthreads();

  // scores: wave w handles k tiles [w*16, w*16+16); ping-pong K fragments
  const float scale = 0.125f;  // 1/sqrt(64)
  {
    auto loadK2 = [&](v16us kb[2], int kt) {
      int kcol = kt * 16 + lo;
#pragma unroll
      for (int c = 0; c < 2; ++c) {
        const unsigned short* kr = K + (size_t)kcol * Dd + c * 32 + hi * 16;
#pragma unroll
        for (int e = 0; e < 16; ++e) kb[c][e] = kr[e];
      }
    };
    auto emit = [&](const v16us kb[2], int kt) {
      v8f acc = {};
      acc = wmma_bf16(qa[0], kb[0], acc);
      acc = wmma_bf16(qa[1], kb[1], acc);
      int kcol = kt * 16 + lo;
      float mval = mask[(size_t)b * Ss + kcol];
#pragma unroll
      for (int r = 0; r < 8; ++r) {
        int row = r + hi * 8;
        int rel = kcol - (q0 + row);
        rel = rel < -64 ? -64 : (rel > 64 ? 64 : rel);
        s_scores[row][kcol] = (acc[r] + s_qtab[row][rel + 64]) * scale + mval;
      }
    };
    int kt0 = wave * 16;
    v16us k0[2], k1[2];
    loadK2(k0, kt0);
    for (int kt = kt0; kt < kt0 + 16; kt += 2) {
      loadK2(k1, kt + 1);
      emit(k0, kt);
      if (kt + 2 < kt0 + 16) loadK2(k0, kt + 2);
      emit(k1, kt + 1);
    }
  }
  __syncthreads();

  // softmax: thread handles (row = tid&15, 128-col chunk = tid>>4)
  {
    int row = tid & 15;
    int ch = tid >> 4;
    int c0 = ch * (Ss / 8);
    float m = -3.4e38f;
    for (int j = 0; j < Ss / 8; ++j) m = fmaxf(m, s_scores[row][c0 + j]);
    s_red[row][ch] = m;
    __syncthreads();
    float mall = s_red[row][0];
#pragma unroll
    for (int c = 1; c < 8; ++c) mall = fmaxf(mall, s_red[row][c]);
    __syncthreads();
    float sum = 0.f;
    for (int j = 0; j < Ss / 8; ++j) {
      float p = __expf(s_scores[row][c0 + j] - mall);
      s_scores[row][c0 + j] = p;
      sum += p;
    }
    s_red[row][ch] = sum;
    __syncthreads();
    float tot = 0.f;
#pragma unroll
    for (int c = 0; c < 8; ++c) tot += s_red[row][c];
    float inv = 1.0f / tot;
    for (int j = 0; j < Ss / 8; ++j) {
      int k = c0 + j;
      float p = s_scores[row][k] * inv;
      s_scores[row][k] = p;
      int rel = k - (q0 + row);
      rel = rel < -64 ? -64 : (rel > 64 ? 64 : rel);
      atomicAdd(&s_pb[row][rel + 64], p);
    }
  }
  __syncthreads();

  // ctx = P @ V : wave w owns D-tile w; ping-pong P/V fragments
  {
    int n = wave;
    auto loadP1 = [&](v16us& pf, int kc) {
#pragma unroll
      for (int e = 0; e < 16; ++e) {
        int k = kc * 32 + hi * 16 + e;
        pf[e] = f32_to_bf16(s_scores[lo][k]);
      }
    };
    auto loadV1 = [&](v16us& vf, int kc) {
      const unsigned short* vr = V + (size_t)(n * 16 + lo) * Ss + kc * 32 + hi * 16;
#pragma unroll
      for (int e = 0; e < 16; ++e) vf[e] = vr[e];
    };
    constexpr int NC = Ss / 32;  // 32 (even)
    v8f cacc = {};
    v16us p0, v0, p1, v1;
    loadP1(p0, 0);
    loadV1(v0, 0);
    for (int kc = 0; kc < NC; kc += 2) {
      loadP1(p1, kc + 1);
      loadV1(v1, kc + 1);
      cacc = wmma_bf16(p0, v0, cacc);
      if (kc + 2 < NC) {
        loadP1(p0, kc + 2);
        loadV1(v0, kc + 2);
      }
      cacc = wmma_bf16(p1, v1, cacc);
    }
#pragma unroll
    for (int r = 0; r < 8; ++r) s_ctx[r + hi * 8][n * 16 + lo] = cacc[r];
  }
  __syncthreads();

  // out = ctx + pb @ tab  (129x64 fp32, tiny)
  for (int i = 0; i < (16 * Dd) / 128; ++i) {
    int idx = tid + 128 * i;
    int row = idx >> 6;
    int d = idx & 63;
    float acc = s_ctx[row][d];
    for (int u = 0; u < VOCAB; ++u) acc += s_pb[row][u] * tabf[u * Dd + d];
    out[((size_t)(b * Ss + q0 + row)) * Hh + h * Dd + d] = acc;
  }
}

extern "C" void kernel_launch(void* const* d_in, const int* in_sizes, int n_in,
                              void* d_out, int out_size, void* d_ws, size_t ws_size,
                              hipStream_t stream) {
  const float* hidden = (const float*)d_in[0];
  const float* mask   = (const float*)d_in[1];
  const float* Wq     = (const float*)d_in[2];
  const float* bq     = (const float*)d_in[3];
  const float* Wk     = (const float*)d_in[4];
  const float* bk     = (const float*)d_in[5];
  const float* Wv     = (const float*)d_in[6];
  const float* bv     = (const float*)d_in[7];
  float* out = (float*)d_out;
  (void)in_sizes; (void)n_in; (void)out_size; (void)ws_size;

  char* ws = (char*)d_ws;
  size_t off = 0;
  auto salloc = [&](size_t bytes) -> char* {
    char* p = ws + off;
    off += (bytes + 255) & ~(size_t)255;
    return p;
  };
  unsigned short* hid_bf = (unsigned short*)salloc((size_t)Bb * Ss * Hh * 2);
  unsigned short* w_bf   = (unsigned short*)salloc((size_t)3 * Hh * Hh * 2);
  unsigned short* qkv_bf = (unsigned short*)salloc((size_t)3 * Bb * NHh * Ss * Dd * 2);
  float*          tabf   = (float*)salloc((size_t)VOCAB * Dd * 4);
  unsigned short* tabbf  = (unsigned short*)salloc((size_t)VPAD * Dd * 2);

  cvt_f32_bf16<<<2048, 256, 0, stream>>>(hidden, hid_bf, Bb * Ss * Hh);
  cvt_f32_bf16<<<1024, 256, 0, stream>>>(Wq, w_bf + 0 * (size_t)Hh * Hh, Hh * Hh);
  cvt_f32_bf16<<<1024, 256, 0, stream>>>(Wk, w_bf + 1 * (size_t)Hh * Hh, Hh * Hh);
  cvt_f32_bf16<<<1024, 256, 0, stream>>>(Wv, w_bf + 2 * (size_t)Hh * Hh, Hh * Hh);
  build_tab<<<36, 256, 0, stream>>>(tabf, tabbf);

  qkv_gemm<<<dim3(256, 1, 3), 128, 0, stream>>>(
      hid_bf, w_bf, w_bf + (size_t)Hh * Hh, w_bf + 2 * (size_t)Hh * Hh,
      bq, bk, bv, qkv_bf);

  const unsigned short* qhp = qkv_bf;
  const unsigned short* khp = qkv_bf + (size_t)Bb * NHh * Ss * Dd;
  const unsigned short* vtp = qkv_bf + 2 * (size_t)Bb * NHh * Ss * Dd;
  attn<<<Bb * NHh * (Ss / 16), 128, 0, stream>>>(qhp, khp, vtp, tabf, tabbf,
                                                 mask, out);
}
